// MMLoss_962072674671
// MI455X (gfx1250) — matrix operand order repaired and checked by reading, not compiled, not run on hardware
//
#include <hip/hip_runtime.h>
#include <math.h>

typedef _Float16 v8h  __attribute__((ext_vector_type(8)));
typedef _Float16 v16h __attribute__((ext_vector_type(16)));
typedef float    v8f  __attribute__((ext_vector_type(8)));

constexpr int B_ = 2, C_ = 128, H_ = 384, W_ = 384;
constexpr int HW_ = H_ * W_;
constexpr int NBHW_ = B_ * HW_;
constexpr int NS_ = 4096;
constexpr int KS_ = 16;
constexpr int BORDER_ = 5;
constexpr float NEGINF_ = -3.0e38f;
constexpr float POSINF_ = 3.0e38f;

// ---------------- device helpers ----------------
__device__ __forceinline__ int clampi(int v, int lo, int hi) {
  return v < lo ? lo : (v > hi ? hi : v);
}
__device__ __forceinline__ float borderv(int y, int x) {
  return (y >= BORDER_ && y < H_ - BORDER_ && x >= BORDER_ && x < W_ - BORDER_) ? 1.f : 0.f;
}
__device__ __forceinline__ int refl(int v, int n) {
  if (v < 0) v = -v;
  if (v >= n) v = 2 * n - 2 - v;
  return v;
}
__device__ __forceinline__ unsigned hashu(unsigned x) {
  x ^= x >> 16; x *= 0x7feb352du; x ^= x >> 15; x *= 0x846ca68bu; x ^= x >> 16;
  return x;
}
__device__ void atomicMaxF(float* addr, float v) {
  unsigned* ua = (unsigned*)addr;
  unsigned old = *ua;
  for (;;) {
    if (__uint_as_float(old) >= v) return;
    unsigned assumed = old;
    old = atomicCAS(ua, assumed, __float_as_uint(v));
    if (old == assumed) return;
  }
}
__device__ void atomicMinF(float* addr, float v) {
  unsigned* ua = (unsigned*)addr;
  unsigned old = *ua;
  for (;;) {
    if (__uint_as_float(old) <= v) return;
    unsigned assumed = old;
    old = atomicCAS(ua, assumed, __float_as_uint(v));
    if (old == assumed) return;
  }
}

// ---------------- small utility kernels ----------------
__global__ void k_fill(float* p, float v, int n) {
  int i = blockIdx.x * blockDim.x + threadIdx.x;
  if (i < n) p[i] = v;
}
__global__ void k_set3(float* acc, int i0, float v0, int i1, float v1, int i2, float v2) {
  if (threadIdx.x == 0 && blockIdx.x == 0) { acc[i0] = v0; acc[i1] = v1; acc[i2] = v2; }
}
__global__ void k_init_edge(float* acc) {
  if (threadIdx.x || blockIdx.x) return;
  acc[13] = POSINF_; acc[14] = NEGINF_;
  acc[15] = POSINF_; acc[16] = NEGINF_;
  acc[17] = POSINF_; acc[18] = NEGINF_;
  acc[19] = POSINF_; acc[20] = NEGINF_;
}

// ---------------- remap / mask kernels ----------------
// bilinear remap: out = remap(src or ones, flow); valid-masked corners
__global__ void k_warp_bilinear(const float* __restrict__ src, const float* __restrict__ flow,
                                float* __restrict__ out, int use_ones) {
  int i = blockIdx.x * blockDim.x + threadIdx.x;
  if (i >= NBHW_) return;
  int b = i / HW_;
  float mx = flow[(size_t)i * 2 + 0];
  float my = flow[(size_t)i * 2 + 1];
  float px = mx * ((float)W_ / (W_ - 1.f)) - 0.5f;
  float py = my * ((float)H_ / (H_ - 1.f)) - 0.5f;
  float x0f = floorf(px), y0f = floorf(py);
  float wx = px - x0f, wy = py - y0f;
  int x0 = (int)x0f, y0 = (int)y0f;
  float v = 0.f;
#pragma unroll
  for (int dy = 0; dy < 2; ++dy) {
#pragma unroll
    for (int dx = 0; dx < 2; ++dx) {
      int yy = y0 + dy, xx = x0 + dx;
      float w = (dx ? wx : 1.f - wx) * (dy ? wy : 1.f - wy);
      if (yy >= 0 && yy < H_ && xx >= 0 && xx < W_) {
        float s = use_ones ? 1.f : src[(size_t)b * HW_ + yy * W_ + xx];
        v += w * s;
      }
    }
  }
  out[i] = v;
}

// gm = (avg3(gmpre) > 0.5) * border
__global__ void k_gm_fin(const float* __restrict__ gmpre, float* __restrict__ gm) {
  int i = blockIdx.x * blockDim.x + threadIdx.x;
  if (i >= NBHW_) return;
  int b = i / HW_, p = i % HW_, y = p / W_, x = p % W_;
  float s = 0.f;
  for (int dy = -1; dy <= 1; ++dy)
    for (int dx = -1; dx <= 1; ++dx) {
      int yy = y + dy, xx = x + dx;
      if (yy >= 0 && yy < H_ && xx >= 0 && xx < W_) s += gmpre[(size_t)b * HW_ + yy * W_ + xx];
    }
  s *= (1.f / 9.f);
  gm[i] = (s > 0.5f ? 1.f : 0.f) * borderv(y, x);
}

// generic avg / max pools (odd k), zero-pad avg, -inf-pad max
__global__ void k_avgpool(const float* __restrict__ in, float* __restrict__ out, int k) {
  int i = blockIdx.x * blockDim.x + threadIdx.x;
  if (i >= NBHW_) return;
  int b = i / HW_, p = i % HW_, y = p / W_, x = p % W_;
  int pad = k / 2;
  float s = 0.f;
  for (int dy = -pad; dy <= pad; ++dy)
    for (int dx = -pad; dx <= pad; ++dx) {
      int yy = y + dy, xx = x + dx;
      if (yy >= 0 && yy < H_ && xx >= 0 && xx < W_) s += in[(size_t)b * HW_ + yy * W_ + xx];
    }
  out[i] = s / (float)(k * k);
}
__global__ void k_maxpool(const float* __restrict__ in, float* __restrict__ out, int k) {
  int i = blockIdx.x * blockDim.x + threadIdx.x;
  if (i >= NBHW_) return;
  int b = i / HW_, p = i % HW_, y = p / W_, x = p % W_;
  int pad = k / 2;
  float m = NEGINF_;
  for (int dy = -pad; dy <= pad; ++dy)
    for (int dx = -pad; dx <= pad; ++dx) {
      int yy = y + dy, xx = x + dx;
      if (yy >= 0 && yy < H_ && xx >= 0 && xx < W_) m = fmaxf(m, in[(size_t)b * HW_ + yy * W_ + xx]);
    }
  out[i] = m;
}

// ---------------- sampler ----------------
__global__ void k_sample(const float* __restrict__ feat1, const float* __restrict__ feat2,
                         const float* __restrict__ flow,
                         const float* __restrict__ s1src, const float* __restrict__ s2src,
                         int b, unsigned seed,
                         _Float16* __restrict__ F1h, _Float16* __restrict__ F2h,
                         float* __restrict__ s1v, float* __restrict__ s2v,
                         float* __restrict__ posx, float* __restrict__ posy) {
  const int s = blockIdx.x;
  const int ch = threadIdx.x;
  const unsigned idx = hashu((unsigned)s * 2654435761u + seed * 0x9e3779b9u + 1u) % (unsigned)HW_;
  const int row = (int)(idx / W_), col = (int)(idx % W_);
  const float mx = flow[((size_t)b * HW_ + idx) * 2 + 0];
  const float my = flow[((size_t)b * HW_ + idx) * 2 + 1];
  const float px = mx * ((float)W_ / (W_ - 1.f)) - 0.5f;
  const float py = my * ((float)H_ / (H_ - 1.f)) - 0.5f;
  const int xi = (int)roundf(px), yi = (int)roundf(py);
  const bool valid = (xi >= 0 && xi < W_ && yi >= 0 && yi < H_);
  const int xc = clampi(xi, 0, W_ - 1), yc = clampi(yi, 0, H_ - 1);
  F1h[(size_t)s * C_ + ch] = (_Float16)feat1[(size_t)(b * C_ + ch) * HW_ + idx];
  F2h[(size_t)s * C_ + ch] =
      (_Float16)(valid ? feat2[(size_t)(b * C_ + ch) * HW_ + yc * W_ + xc] : 0.f);
  if (ch == 0) {
    s1v[s] = s1src[(size_t)b * HW_ + idx];
    s2v[s] = s2src[(size_t)b * HW_ + idx];
    posx[s] = (float)col;
    posy[s] = (float)row;
  }
}

// ---------------- WMMA GEMM + fused masked row/col max epilogue ----------------
// S = A(4096x128) @ B(4096x128)^T, f16 inputs, f32 accum, 16x16 tile per wave,
// K in 4 chunks of 32 via v_wmma_f32_16x16x32_f16. B tile staged in LDS once per
// block (shared by the 8 waves). Epilogue applies hard_dist masks and reduces
// into per-column / per-row max vectors with CAS float-atomic-max.
// mode 0: simi -> colP (p candidates), colC/rowC (neg_c candidates)
// mode 1/2: self-similarity -> colC only (neg_k / neg_j)
__global__ __launch_bounds__(256) void k_gemm(
    const _Float16* __restrict__ A, const _Float16* __restrict__ Bm,
    const float* __restrict__ posx, const float* __restrict__ posy,
    float* colC, float* colP, float* rowC, int mode) {
  __shared__ alignas(32) _Float16 ldsB[16 * C_];
  const int t = threadIdx.x;
  const int wave = t >> 5;
  const int lane = t & 31;
  const int n0 = blockIdx.x * 16;
  const int m0 = (blockIdx.y * 8 + wave) * 16;

  // stage B tile (16 rows x 128 halves, contiguous) into LDS
  *(v8h*)(ldsB + t * 8) = *(const v8h*)(Bm + (size_t)n0 * C_ + t * 8);
  __syncthreads();

  const int mrow = m0 + (lane & 15);
  const int ahalf = (lane >> 4) * 16;  // A frag: low lanes K 0..15, high lanes 16..31
  const int nloc = lane & 15;
  const int bhalf = (lane >> 4) * 16;  // B frag: low lanes K 0..15, high lanes 16..31
  const _Float16* arow = A + (size_t)mrow * C_;
  __builtin_prefetch(arow, 0, 1);

  v8f c = {};
#pragma unroll
  for (int k0 = 0; k0 < C_; k0 += 32) {
    v8h alo = *(const v8h*)(arow + k0 + ahalf);
    v8h ahi = *(const v8h*)(arow + k0 + ahalf + 8);
    v16h a;
#pragma unroll
    for (int e = 0; e < 8; ++e) { a[e] = alo[e]; a[e + 8] = ahi[e]; }
    v16h bfrag = *(const v16h*)(ldsB + nloc * C_ + k0 + bhalf);
    c = __builtin_amdgcn_wmma_f32_16x16x32_f16(false, a, false, bfrag, (short)0, c,
                                               false, false);
  }

  // epilogue: masked max reductions (C/D layout: VGPR r -> M = r (+8 for hi lanes))
  const int n = n0 + nloc;
  const float pxn = posx[n], pyn = posy[n];
  float cmaxC = NEGINF_, cmaxP = NEGINF_;
  float rloc[8];
#pragma unroll
  for (int r = 0; r < 8; ++r) {
    const int m = m0 + r + ((lane >> 4) << 3);
    const float s = c[r];
    const float dx = posx[m] - pxn, dy = posy[m] - pyn;
    const float d2 = dx * dx + dy * dy;
    const float sm = s - (s > 0.9f ? 10.f : 0.f) - (d2 < 49.f ? 10.f : 0.f);
    cmaxC = fmaxf(cmaxC, sm);
    rloc[r] = sm;
    if (mode == 0) cmaxP = fmaxf(cmaxP, s - (d2 < 9.f ? 0.f : 10.f));
  }
  atomicMaxF(&colC[n], cmaxC);
  if (mode == 0) {
    atomicMaxF(&colP[n], cmaxP);
#pragma unroll
    for (int r = 0; r < 8; ++r) {
      float v = rloc[r];
      for (int off = 1; off < 16; off <<= 1) v = fmaxf(v, __shfl_xor(v, off, 32));
      if ((lane & 15) == 0) atomicMaxF(&rowC[m0 + r + ((lane >> 4) << 3)], v);
    }
  }
}

// ---------------- hard_dist combine ----------------
__global__ void k_hdist(const float* __restrict__ colP, const float* __restrict__ colC,
                        const float* __restrict__ rowC, const float* __restrict__ Kb,
                        const float* __restrict__ Jb, float* __restrict__ M) {
  int j = blockIdx.x * blockDim.x + threadIdx.x;
  if (j >= NS_) return;
  const float lo = -1.f + 1e-5f, hi = 1.f - 1e-5f;
  float p = fminf(fmaxf(colP[j], lo), hi);
  float nc = fmaxf(fminf(fmaxf(colC[j], lo), hi), fminf(fmaxf(rowC[j], lo), hi));
  float nk = fminf(fmaxf(Kb[j], lo), hi);
  float nj = fminf(fmaxf(Jb[j], lo), hi);
  const float pi = 3.14159265358979f;
  float a = pi - acosf(nk), b = pi - acosf(nj), cc = pi - acosf(nc), d = acosf(p);
  float s = a * a * (1.f / 3.f) + b * b * (1.f / 3.f) + cc * cc * (1.f / 3.f) + d * d;
  M[j] = s * s;
}

__global__ void k_desc_acc(const float* __restrict__ M, const float* __restrict__ s1v,
                           const float* __restrict__ s2v, float* acc, int b) {
  int j = blockIdx.x * blockDim.x + threadIdx.x;
  if (j >= NS_) return;
  float sc = s1v[j] * s2v[j];
  atomicAdd(&acc[b], sc * M[j]);
  atomicAdd(&acc[2 + b], sc);
}

__global__ void k_M_stats(const float* __restrict__ M, float* acc) {
  int j = blockIdx.x * blockDim.x + threadIdx.x;
  if (j >= NS_) return;
  float v = M[j];
  atomicMinF(&acc[10], v);
  atomicMaxF(&acc[11], v);
  atomicAdd(&acc[12], v);
}

__global__ void k_coup(const float* __restrict__ M, const float* __restrict__ s1v,
                       const float* __restrict__ s2v, float* acc, int b) {
  int j = blockIdx.x * blockDim.x + threadIdx.x;
  if (j >= NS_) return;
  float mn = acc[10], mx = acc[11];
  float rng = fmaxf(mx - mn, 1e-12f);
  float meanN = (acc[12] / (float)NS_ - mn) / rng;
  float Mn = (M[j] - mn) / rng;
  float mask3 = fmaxf(0.f, 1.f - Mn / (meanN + 1e-12f));
  float a = 1.f - s1v[j], b2 = 1.f - s2v[j];
  atomicAdd(&acc[4 + b], mask3 * a * a);
  atomicAdd(&acc[6 + b], mask3 * b2 * b2);
  atomicAdd(&acc[8 + b], mask3);
}

// ---------------- edge loss pipeline ----------------
__global__ void k_grad(const float* __restrict__ img, float* __restrict__ g) {
  int i = blockIdx.x * blockDim.x + threadIdx.x;
  if (i >= NBHW_) return;
  int b = i / HW_, p = i % HW_, y = p / W_, x = p % W_;
  float accv = 0.f;
  for (int ch = 0; ch < 3; ++ch) {
    const float* im = img + (size_t)(b * 3 + ch) * HW_;
    float v[3][3];
    for (int dy = -1; dy <= 1; ++dy)
      for (int dx = -1; dx <= 1; ++dx)
        v[dy + 1][dx + 1] = im[clampi(y + dy, 0, H_ - 1) * W_ + clampi(x + dx, 0, W_ - 1)];
    float gxx = (v[0][0] - 2.f * v[0][1] + v[0][2] + 2.f * v[1][0] - 4.f * v[1][1] +
                 2.f * v[1][2] + v[2][0] - 2.f * v[2][1] + v[2][2]) * (1.f / 16.f);
    float gxy = (-v[0][0] + v[0][2] + v[2][0] - v[2][2]) * 0.25f;
    float gyy = (v[0][0] + 2.f * v[0][1] + v[0][2] - 2.f * v[1][0] - 4.f * v[1][1] -
                 2.f * v[1][2] + v[2][0] + 2.f * v[2][1] + v[2][2]) * (1.f / 16.f);
    accv += fabsf(gxx) + fabsf(gxy) + fabsf(gyy);
  }
  g[i] = accv;
}

__global__ void k_minmax(const float* __restrict__ e, float* acc, int base) {
  int i = blockIdx.x * blockDim.x + threadIdx.x;
  if (i >= NBHW_) return;
  int b = i / HW_;
  atomicMinF(&acc[base + b * 2], e[i]);
  atomicMaxF(&acc[base + b * 2 + 1], e[i]);
}

__global__ void k_prsum(const float* __restrict__ e, float* acc, int base, int slot) {
  int i = blockIdx.x * blockDim.x + threadIdx.x;
  if (i >= NBHW_) return;
  int b = i / HW_;
  float mn = acc[base + b * 2], mx = acc[base + b * 2 + 1];
  float pr = (e[i] - mn) / fmaxf(mx - mn, 1e-20f);
  atomicAdd(&acc[slot], pr);
}

__global__ void k_edge_loss(const float* __restrict__ e, const float* __restrict__ score,
                            float* acc, int base, int prslot, int slot) {
  int i = blockIdx.x * blockDim.x + threadIdx.x;
  if (i >= NBHW_) return;
  int b = i / HW_, p = i % HW_, y = p / W_, x = p % W_;
  float mn = acc[base + b * 2], mx = acc[base + b * 2 + 1];
  float pr = (e[i] - mn) / fmaxf(mx - mn, 1e-20f);
  float mean = acc[prslot] / (float)NBHW_;
  float mask = fmaxf(0.f, 1.f - pr / (mean + 1e-12f));
  float sc = score[i] * borderv(y, x);
  atomicAdd(&acc[slot], mask * sc * sc);
}

// ---------------- gaussian blur + peak loss ----------------
__global__ void k_gauss3(const float* __restrict__ score, float* __restrict__ out) {
  int i = blockIdx.x * blockDim.x + threadIdx.x;
  if (i >= NBHW_) return;
  int b = i / HW_, p = i % HW_, y = p / W_, x = p % W_;
  const float w0 = 0.27406862f, w1 = 0.45186276f;
  const float wv[3] = {w0, w1, w0};
  float s = 0.f;
  for (int dy = -1; dy <= 1; ++dy)
    for (int dx = -1; dx <= 1; ++dx) {
      int ry = refl(y + dy, H_), rx = refl(x + dx, W_);
      float val = score[(size_t)b * HW_ + ry * W_ + rx] * borderv(ry, rx);
      s += wv[dy + 1] * wv[dx + 1] * val;
    }
  out[i] = s;
}

__global__ void k_peak(const float* __restrict__ sb, float* acc, int slot) {
  int i = blockIdx.x * blockDim.x + threadIdx.x;
  if (i >= NBHW_) return;
  int b = i / HW_, p = i % HW_, y = p / W_, x = p % W_;
  const float* s = sb + (size_t)b * HW_;
  float s17 = 0.f, m17 = NEGINF_;
  for (int dy = -8; dy <= 8; ++dy)
    for (int dx = -8; dx <= 8; ++dx) {
      int yy = y + dy, xx = x + dx;
      if (yy >= 0 && yy < H_ && xx >= 0 && xx < W_) {
        float v = s[yy * W_ + xx];
        s17 += v;
        m17 = fmaxf(m17, v);
      }
    }
  float a17 = s17 / 289.f;
  float s3 = 0.f, m3 = NEGINF_;
  for (int dy = -1; dy <= 1; ++dy)
    for (int dx = -1; dx <= 1; ++dx) {
      int yy = y + dy, xx = x + dx;
      if (yy >= 0 && yy < H_ && xx >= 0 && xx < W_) {
        float v = s[yy * W_ + xx];
        s3 += v;
        m3 = fmaxf(m3, v);
      }
    }
  float a3 = s3 / 9.f;
  float u = 1.f - m17, w = a3 + 1.f - m3;
  atomicAdd(&acc[slot], a17 * a17 + u * u + w * w);
}

// ---------------- repeatability loss (unfold patches) ----------------
__global__ __launch_bounds__(256) void k_rep(
    const float* __restrict__ feat1, const float* __restrict__ feat2,
    const float* __restrict__ flow, const float* __restrict__ score1,
    const float* __restrict__ s2w, const float* __restrict__ gm, float* acc, int slot) {
  __shared__ float r0[256], r1[256], r2[256], r3[256];
  int b = blockIdx.z;
  int py = blockIdx.y, px = blockIdx.x;
  int t = threadIdx.x;
  int ty = t / KS_, tx = t % KS_;
  int y = py * (KS_ / 2) + ty, x = px * (KS_ / 2) + tx;
  size_t pix = (size_t)b * HW_ + y * W_ + x;
  float gmv = gm[pix];
  float mx = flow[pix * 2 + 0], my = flow[pix * 2 + 1];
  float pxu = mx * ((float)W_ / (W_ - 1.f)) - 0.5f;
  float pyu = my * ((float)H_ / (H_ - 1.f)) - 0.5f;
  int xi = (int)roundf(pxu), yi = (int)roundf(pyu);
  bool valid = (xi >= 0 && xi < W_ && yi >= 0 && yi < H_);
  int xc = clampi(xi, 0, W_ - 1), yc = clampi(yi, 0, H_ - 1);
  float dot = 0.f;
  if (valid && gmv > 0.f) {
    const float* f1 = feat1 + (size_t)b * C_ * HW_ + y * W_ + x;
    const float* f2 = feat2 + (size_t)b * C_ * HW_ + yc * W_ + xc;
    for (int ch = 0; ch < C_; ++ch) dot += f1[(size_t)ch * HW_] * f2[(size_t)ch * HW_];
  }
  float fps = dot * gmv;
  float a = score1[pix] * gmv;
  float c2 = s2w[pix] * gmv;
  r0[t] = fps; r1[t] = a * a; r2[t] = c2 * c2; r3[t] = a * c2;
  __syncthreads();
  for (int off = 128; off > 0; off >>= 1) {
    if (t < off) { r0[t] += r0[t + off]; r1[t] += r1[t + off]; r2[t] += r2[t + off]; r3[t] += r3[t + off]; }
    __syncthreads();
  }
  if (t == 0) {
    float ssum = r0[0];
    float na = fmaxf(sqrtf(r1[0]), 1e-12f), nb = fmaxf(sqrtf(r2[0]), 1e-12f);
    float cosim = r3[0] / (na * nb);
    float psimi = fmaxf(ssum, 0.f) / (ssum + 1.f);
    atomicAdd(&acc[slot], psimi * (1.f - cosim));
  }
}

// ---------------- final combine (sequential rss / m3m recurrences) ----------------
__global__ void k_finalize(const float* __restrict__ acc, float* __restrict__ out) {
  if (threadIdx.x || blockIdx.x) return;
  float rss = 1000.f, ld = 0.f;
  for (int b = 0; b < B_; ++b) {
    ld += acc[b] / (rss + 1e-5f);
    rss = 0.99f * rss + 0.01f * acc[2 + b];
  }
  ld /= (float)B_;
  float m3m = 10.f, lc = 0.f;
  for (int b = 0; b < B_; ++b) {
    lc += (acc[4 + b] / (float)NS_ + acc[6 + b] / (float)NS_) / m3m;
    m3m = 0.99f * m3m + 0.01f * (acc[8 + b] / (float)NS_);
  }
  lc /= (float)B_;
  const float inv = 1.f / (float)NBHW_;
  float ledge = acc[23] * inv * 0.1f + acc[24] * inv * 0.1f;
  float lrand = acc[25] * inv + acc[26] * inv;
  float lrep = acc[27] / (100.f + 1e-5f);
  out[0] = ld + (ledge + lrand + lc) + lrep;
}

// ---------------- host launch ----------------
extern "C" void kernel_launch(void* const* d_in, const int* in_sizes, int n_in,
                              void* d_out, int out_size, void* d_ws, size_t ws_size,
                              hipStream_t stream) {
  (void)in_sizes; (void)n_in; (void)out_size; (void)ws_size;
  const float* feat1 = (const float*)d_in[0];
  const float* score1 = (const float*)d_in[1];
  const float* feat2 = (const float*)d_in[2];
  const float* score2 = (const float*)d_in[3];
  const float* flow = (const float*)d_in[4];
  const float* img1 = (const float*)d_in[5];
  const float* img2 = (const float*)d_in[6];
  float* out = (float*)d_out;

  // workspace layout
  float* p = (float*)d_ws;
  float* acc = p;      p += 64;
  float* gmpre = p;    p += NBHW_;
  float* gm = p;       p += NBHW_;
  float* s2w = p;      p += NBHW_;
  float* s1a = p;      p += NBHW_;
  float* s2a = p;      p += NBHW_;
  float* g = p;        p += NBHW_;
  float* tA = p;       p += NBHW_;
  float* e1 = p;       p += NBHW_;
  float* e2 = p;       p += NBHW_;
  float* sc1b = p;     p += NBHW_;
  float* sc2b = p;     p += NBHW_;
  float* colP = p;
  float* colC = colP + NS_;
  float* rowC = colC + NS_;
  float* Kb = rowC + NS_;
  float* Jb = Kb + NS_;        p += 5 * NS_;
  float* Mv = p;       p += NS_;
  float* s1v = p;      p += NS_;
  float* s2v = p;      p += NS_;
  float* posx = p;     p += NS_;
  float* posy = p;     p += NS_;
  _Float16* F1h = (_Float16*)p;  p += NS_ * C_ / 2;
  _Float16* F2h = (_Float16*)p;  p += NS_ * C_ / 2;

  const int TPB = 256;
  const int NB = (NBHW_ + TPB - 1) / TPB;
  const dim3 ggrid(NS_ / 16, NS_ / 16 / 8);   // 256 x 32 blocks, 8 waves each
  const int NSB = NS_ / TPB;

  hipMemsetAsync(acc, 0, 64 * sizeof(float), stream);
  k_init_edge<<<1, 1, 0, stream>>>(acc);

  // warps / masks / pooled scores
  k_warp_bilinear<<<NB, TPB, 0, stream>>>(score2, flow, gmpre, 1);
  k_gm_fin<<<NB, TPB, 0, stream>>>(gmpre, gm);
  k_warp_bilinear<<<NB, TPB, 0, stream>>>(score2, flow, s2w, 0);
  k_avgpool<<<NB, TPB, 0, stream>>>(score1, s1a, 3);
  k_avgpool<<<NB, TPB, 0, stream>>>(s2w, s2a, 3);

  // loss_desc: sampler(seed b) -> 3 WMMA GEMMs -> hard_dist -> accumulate
  for (int b = 0; b < B_; ++b) {
    k_sample<<<NS_, C_, 0, stream>>>(feat1, feat2, flow, s1a, s2a, b, (unsigned)b,
                                     F1h, F2h, s1v, s2v, posx, posy);
    k_fill<<<(5 * NS_ + TPB - 1) / TPB, TPB, 0, stream>>>(colP, NEGINF_, 5 * NS_);
    k_gemm<<<ggrid, TPB, 0, stream>>>(F1h, F2h, posx, posy, colC, colP, rowC, 0);
    k_gemm<<<ggrid, TPB, 0, stream>>>(F1h, F1h, posx, posy, Kb, colP, rowC, 1);
    k_gemm<<<ggrid, TPB, 0, stream>>>(F2h, F2h, posx, posy, Jb, colP, rowC, 2);
    k_hdist<<<NSB, TPB, 0, stream>>>(colP, colC, rowC, Kb, Jb, Mv);
    k_desc_acc<<<NSB, TPB, 0, stream>>>(Mv, s1v, s2v, acc, b);
  }

  // loss_coup: sampler(seed 100+b) with raw scores
  for (int b = 0; b < B_; ++b) {
    k_sample<<<NS_, C_, 0, stream>>>(feat1, feat2, flow, score1, s2w, b,
                                     (unsigned)(100 + b), F1h, F2h, s1v, s2v, posx, posy);
    k_fill<<<(5 * NS_ + TPB - 1) / TPB, TPB, 0, stream>>>(colP, NEGINF_, 5 * NS_);
    k_gemm<<<ggrid, TPB, 0, stream>>>(F1h, F2h, posx, posy, colC, colP, rowC, 0);
    k_gemm<<<ggrid, TPB, 0, stream>>>(F1h, F1h, posx, posy, Kb, colP, rowC, 1);
    k_gemm<<<ggrid, TPB, 0, stream>>>(F2h, F2h, posx, posy, Jb, colP, rowC, 2);
    k_hdist<<<NSB, TPB, 0, stream>>>(colP, colC, rowC, Kb, Jb, Mv);
    k_set3<<<1, 1, 0, stream>>>(acc, 10, POSINF_, 11, NEGINF_, 12, 0.f);
    k_M_stats<<<NSB, TPB, 0, stream>>>(Mv, acc);
    k_coup<<<NSB, TPB, 0, stream>>>(Mv, s1v, s2v, acc, b);
  }

  // edge masks + loss (img1 with score1, img2 with score2)
  k_grad<<<NB, TPB, 0, stream>>>(img1, g);
  k_maxpool<<<NB, TPB, 0, stream>>>(g, tA, 7);
  k_avgpool<<<NB, TPB, 0, stream>>>(tA, e1, 3);
  k_minmax<<<NB, TPB, 0, stream>>>(e1, acc, 13);
  k_prsum<<<NB, TPB, 0, stream>>>(e1, acc, 13, 21);
  k_edge_loss<<<NB, TPB, 0, stream>>>(e1, score1, acc, 13, 21, 23);

  k_grad<<<NB, TPB, 0, stream>>>(img2, g);
  k_maxpool<<<NB, TPB, 0, stream>>>(g, tA, 7);
  k_avgpool<<<NB, TPB, 0, stream>>>(tA, e2, 3);
  k_minmax<<<NB, TPB, 0, stream>>>(e2, acc, 17);
  k_prsum<<<NB, TPB, 0, stream>>>(e2, acc, 17, 22);
  k_edge_loss<<<NB, TPB, 0, stream>>>(e2, score2, acc, 17, 22, 24);

  // gaussian blur + peak_rand
  k_gauss3<<<NB, TPB, 0, stream>>>(score1, sc1b);
  k_gauss3<<<NB, TPB, 0, stream>>>(score2, sc2b);
  k_peak<<<NB, TPB, 0, stream>>>(sc1b, acc, 25);
  k_peak<<<NB, TPB, 0, stream>>>(sc2b, acc, 26);

  // repeatability (unfold patches, stride KS/2, 47x47 patches)
  {
    dim3 rgrid((W_ - KS_) / (KS_ / 2) + 1, (H_ - KS_) / (KS_ / 2) + 1, B_);
    k_rep<<<rgrid, TPB, 0, stream>>>(feat1, feat2, flow, score1, s2w, gm, acc, 27);
  }

  k_finalize<<<1, 1, 0, stream>>>(acc, out);
}
